// complex_UNet4D_61426622267628
// MI455X (gfx1250) — matrix operand order repaired and checked by reading, not compiled, not run on hardware
//
#include <hip/hip_runtime.h>

// CDNA5 / gfx1250 implicit-GEMM 4D U-Net, bf16 end-to-end.
//  - Activations: bf16, channel-minor [n][C] (row stride ldC, channel base cb)
//    -> B fragment = 2x global_load_b128 per lane, concat = free (slice writes).
//  - Weights: prepacked once per call into exact A-fragment order (bf16,
//    zero-padded over Co) -> A fragment = 2x global_load_b128, no in-loop cvt.
//  - 2x2 register macro-tile (32x32 output / wave32), 4x
//    v_wmma_f32_16x16x32_bf16 per 32-wide K chunk, f32 accumulate.
//  - Epilogue: bias+ReLU fused, packed b128 bf16 store (channel-minor);
//    the f32 head epilogue is a separate template instantiation (cold path).

typedef __attribute__((ext_vector_type(16))) __bf16   v16bf;
typedef __attribute__((ext_vector_type(8)))  unsigned v8u;
typedef __attribute__((ext_vector_type(4)))  unsigned v4u;
typedef __attribute__((ext_vector_type(8)))  float    v8f;
typedef __attribute__((ext_vector_type(4)))  float    v4f;
typedef unsigned short ush;

struct ConvP {
  int Ci, Co;
  int Ti, Di, Hi, Wi;
  int To, Do, Ho, Wo;
  int kT, kD, kH, kW;
  int s, p, relu;
  int ldIn, cbIn, ldOut, cbOut;   // channel-minor row strides / channel bases
  int NC, MT;                     // NC = Ci/32 chunks, MT = #16-row m-tiles (padded)
};

// Pack two f32 -> two bf16 (round-to-nearest) in one dword.
__device__ __forceinline__ unsigned pk(float lo, float hi) {
  unsigned ulo = __float_as_uint(lo) + 0x8000u;
  unsigned uhi = __float_as_uint(hi) + 0x8000u;
#if __has_builtin(__builtin_amdgcn_perm)
  return __builtin_amdgcn_perm(uhi, ulo, 0x07060302u);
#else
  return (ulo >> 16) | (uhi & 0xFFFF0000u);
#endif
}

__device__ __forceinline__ void dec_n(int n, const ConvP& P,
                                      int& to, int& od, int& oh, int& ow) {
  int r = n;
  ow = r % P.Wo; r /= P.Wo;
  oh = r % P.Ho; r /= P.Ho;
  od = r % P.Do; to = r / P.Do;
}

__device__ __forceinline__ v8f wmma_bf16(v8u a, v8u b, v8f c) {
  return __builtin_amdgcn_wmma_f32_16x16x32_bf16(
      false, __builtin_bit_cast(v16bf, a),
      false, __builtin_bit_cast(v16bf, b), (short)0, c, false, false);
}

// Fused epilogue for one 8-row quadrant. rbase includes the 8*g term.
template <bool OUTF32>
__device__ __forceinline__ void epi_tile(const v8f& acc, int rbase, int nq, int nvq,
                                         const float* __restrict__ bias,
                                         void* __restrict__ yout,
                                         const ConvP& P, int Nout) {
  if constexpr (!OUTF32) {
    if (nvq) {
      const v4f* bp = (const v4f*)(bias + rbase);
      const v4f b0 = bp[0], b1 = bp[1];
      float v[8];
#pragma unroll
      for (int r = 0; r < 8; ++r) {
        float t = acc[r] + (r < 4 ? b0[r] : b1[r - 4]);
        v[r] = P.relu ? fmaxf(t, 0.f) : t;
      }
      v4u d;
#pragma unroll
      for (int t = 0; t < 4; ++t) d[t] = pk(v[2 * t], v[2 * t + 1]);
      *(v4u*)((ush*)yout + nq * P.ldOut + P.cbOut + rbase) = d;
    }
  } else {                       // final 1x1 head: f32 channel-major output
#pragma unroll
    for (int r = 0; r < 8; ++r) {
      const int co = rbase + r;
      if (co < P.Co && nvq) {
        float t = acc[r] + bias[co];
        if (P.relu) t = fmaxf(t, 0.f);
        ((float*)yout)[co * Nout + nq] = t;
      }
    }
  }
}

// ======================= main conv kernel (Ci%32==0) =======================
template <bool TRANS, bool OUTF32>
__global__ __launch_bounds__(256)
void conv_bf16_wmma(const ush* __restrict__ xb, const unsigned* __restrict__ wp,
                    const float* __restrict__ bias, void* __restrict__ yout,
                    ConvP P)
{
  extern __shared__ int s_tab[];
  const int kvol = P.kT * P.kD * P.kH * P.kW;
  for (int kr = (int)threadIdx.x; kr < kvol; kr += blockDim.x) {
    int kw = kr % P.kW; int r = kr / P.kW;
    int kh = r % P.kH;  r /= P.kH;
    int kd = r % P.kD;  int kt = r / P.kD;
    s_tab[kr] = (kt << 12) | (kd << 8) | (kh << 4) | kw;
  }
  __syncthreads();

  const int lane = threadIdx.x & 31;
  const int wave = (int)((blockIdx.x * blockDim.x + threadIdx.x) >> 5);
  const int Nout = P.To * P.Do * P.Ho * P.Wo;
  const int Mt = (P.Co + 31) >> 5, Nt = (Nout + 31) >> 5;
  if (wave >= Mt * Nt) return;               // wave-uniform: EXEC all-1 at WMMA
  const int mt = wave / Nt, nt = wave % Nt;
  const int g = lane >> 4, l = lane & 15;
  const int n0 = nt * 32 + l, n1 = n0 + 16;

  int bt[2], bd[2], bh[2], bw[2], nv[2];
  {
    const int nn[2] = {n0, n1};
#pragma unroll
    for (int q = 0; q < 2; ++q) {
      nv[q] = (nn[q] < Nout);
      int to, od, oh, ow;
      dec_n(nv[q] ? nn[q] : 0, P, to, od, oh, ow);
      if constexpr (!TRANS) {
        bt[q] = to * P.s - P.p; bd[q] = od * P.s - P.p;
        bh[q] = oh * P.s - P.p; bw[q] = ow * P.s - P.p;
      } else {
        bt[q] = to + P.p; bd[q] = od + P.p; bh[q] = oh + P.p; bw[q] = ow + P.p;
      }
    }
  }

  v8f acc00 = {}, acc01 = {}, acc10 = {}, acc11 = {};

  const int nCh = P.NC * kvol;
  int kr = 0, cc = 0;
  for (int ch = 0; ch < nCh; ++ch) {
    const int c  = s_tab[kr];
    const int kt = (c >> 12) & 15, kd = (c >> 8) & 15,
              kh = (c >> 4) & 15, kwv = c & 15;

    int bOff[2]; unsigned mk[2];
#pragma unroll
    for (int q = 0; q < 2; ++q) {
      int it, id, ih, iw, ok;
      if constexpr (!TRANS) {
        it = bt[q] + kt; id = bd[q] + kd; ih = bh[q] + kh; iw = bw[q] + kwv;
        ok = nv[q] & ((unsigned)it < (unsigned)P.Ti) & ((unsigned)id < (unsigned)P.Di)
                   & ((unsigned)ih < (unsigned)P.Hi) & ((unsigned)iw < (unsigned)P.Wi);
      } else {        // stride 2: i = (o + p - k)/2 when >=0 and even
        const int at = bt[q] - kt, ad = bd[q] - kd, ah = bh[q] - kh, aw = bw[q] - kwv;
        const int fold = at | ad | ah | aw;
        it = at >> 1; id = ad >> 1; ih = ah >> 1; iw = aw >> 1;
        ok = nv[q] & (fold >= 0) & ((fold & 1) == 0)
                   & (it < P.Ti) & (id < P.Di) & (ih < P.Hi) & (iw < P.Wi);
      }
      const int sidx = ((it * P.Di + id) * P.Hi + ih) * P.Wi + iw;
      bOff[q] = ok * (sidx * P.ldIn + P.cbIn + cc * 32 + g * 16);  // branch-free
      mk[q]   = ok ? 0xFFFFFFFFu : 0u;                              // bf16 0 == 0x0000
    }
    const v4u m0 = {mk[0], mk[0], mk[0], mk[0]};
    const v4u m1 = {mk[1], mk[1], mk[1], mk[1]};

    // A: prepacked fragment order: ((((kr*NC+cc)*MT + mt16)*2 + g)*16 + l)*8
    const int abase = (kr * P.NC + cc) * P.MT + 2 * mt;
    const int aO0 = ((abase * 2 + g) * 16 + l) * 8;
    const int aO1 = (((abase + 1) * 2 + g) * 16 + l) * 8;

    const v4u a0l = *(const v4u*)(wp + aO0), a0h = *(const v4u*)(wp + aO0 + 4);
    const v4u a1l = *(const v4u*)(wp + aO1), a1h = *(const v4u*)(wp + aO1 + 4);
    const v4u b0l = *(const v4u*)(xb + bOff[0])       & m0;
    const v4u b0h = *((const v4u*)(xb + bOff[0]) + 1) & m0;
    const v4u b1l = *(const v4u*)(xb + bOff[1])       & m1;
    const v4u b1h = *((const v4u*)(xb + bOff[1]) + 1) & m1;

    const v8u A0 = __builtin_shufflevector(a0l, a0h, 0, 1, 2, 3, 4, 5, 6, 7);
    const v8u A1 = __builtin_shufflevector(a1l, a1h, 0, 1, 2, 3, 4, 5, 6, 7);
    const v8u B0 = __builtin_shufflevector(b0l, b0h, 0, 1, 2, 3, 4, 5, 6, 7);
    const v8u B1 = __builtin_shufflevector(b1l, b1h, 0, 1, 2, 3, 4, 5, 6, 7);

    acc00 = wmma_bf16(A0, B0, acc00);
    acc01 = wmma_bf16(A0, B1, acc01);
    acc10 = wmma_bf16(A1, B0, acc10);
    acc11 = wmma_bf16(A1, B1, acc11);

    if (++cc == P.NC) { cc = 0; ++kr; }
  }

  const int rb = 2 * mt * 16 + 8 * g;
  epi_tile<OUTF32>(acc00, rb,      n0, nv[0], bias, yout, P, Nout);
  epi_tile<OUTF32>(acc01, rb,      n1, nv[1], bias, yout, P, Nout);
  epi_tile<OUTF32>(acc10, rb + 16, n0, nv[0], bias, yout, P, Nout);
  epi_tile<OUTF32>(acc11, rb + 16, n1, nv[1], bias, yout, P, Nout);
}

// ================= generic 16x16 kernel for c1a (Ci == 1) =================
__global__ __launch_bounds__(256)
void conv_c1a(const ush* __restrict__ xb, const float* __restrict__ w,
              const float* __restrict__ bias, ush* __restrict__ y, ConvP P)
{
  extern __shared__ int s_tab[];
  const int kvol = P.kT * P.kD * P.kH * P.kW;
  const int K = kvol;                          // Ci == 1
  for (int kr = (int)threadIdx.x; kr < kvol; kr += blockDim.x) {
    int kw = kr % P.kW; int r = kr / P.kW;
    int kh = r % P.kH;  r /= P.kH;
    int kd = r % P.kD;  int kt = r / P.kD;
    s_tab[kr] = (kt << 12) | (kd << 8) | (kh << 4) | kw;
  }
  __syncthreads();

  const int lane = threadIdx.x & 31;
  const int wave = (int)((blockIdx.x * blockDim.x + threadIdx.x) >> 5);
  const int Nout = P.To * P.Do * P.Ho * P.Wo;
  const int Mt = (P.Co + 15) >> 4, Nt = (Nout + 15) >> 4;
  if (wave >= Mt * Nt) return;
  const int mt = wave / Nt, nt = wave % Nt;
  const int g = lane >> 4, l = lane & 15;
  const int m = mt * 16 + l;
  const int n = nt * 16 + l;
  const int nvq = (n < Nout);
  int to, od, oh, ow;
  dec_n(nvq ? n : 0, P, to, od, oh, ow);
  const int it0 = to * P.s - P.p, id0 = od * P.s - P.p,
            ih0 = oh * P.s - P.p, iw0 = ow * P.s - P.p;
  const int   mv   = (m < P.Co);
  const int   m_cl = mv ? m : (P.Co - 1);
  const float am   = (float)mv;

  v8f acc = {};
  for (int k0 = 0; k0 < K; k0 += 32) {
    v8u au, bu;
#pragma unroll
    for (int i = 0; i < 8; ++i) {
      float av2[2]; unsigned bv2[2];
#pragma unroll
      for (int h = 0; h < 2; ++h) {
        const int e  = 2 * i + h;
        const int da = e + (e >= 8 ? 8 : 0);
        // ---- A (raw f32 weights, Ci==1: widx = m*kvol + k) ----
        const int ka = k0 + g * 8 + da;
        const int kca = ka < K ? ka : K - 1;
        av2[h] = w[m_cl * kvol + kca] * ((ka < K) ? am : 0.f);
        // ---- B (bf16 activations, spatial index only) ----
        const int kb = k0 + g * 8 + da;
        const int kcb = kb < K ? kb : K - 1;
        const int c = s_tab[kcb];
        const int kt = (c >> 12) & 15, kd = (c >> 8) & 15,
                  kh = (c >> 4) & 15, kwv = c & 15;
        const int it = it0 + kt, id = id0 + kd, ih = ih0 + kh, iw = iw0 + kwv;
        const int ok = nvq & (kb < K)
                     & ((unsigned)it < (unsigned)P.Ti) & ((unsigned)id < (unsigned)P.Di)
                     & ((unsigned)ih < (unsigned)P.Hi) & ((unsigned)iw < (unsigned)P.Wi);
        const int idx = ok * (((it * P.Di + id) * P.Hi + ih) * P.Wi + iw);
        bv2[h] = (unsigned)xb[idx] * (unsigned)ok;
      }
      au[i] = pk(av2[0], av2[1]);
      bu[i] = bv2[0] | (bv2[1] << 16);
    }
    acc = wmma_bf16(au, bu, acc);
  }
  epi_tile<false>(acc, mt * 16 + 8 * g, n, nvq, bias, y, P, Nout);
}

// ================= weight prepack (A-fragment order, bf16) =================
struct PackP { int kvol, NC, MT, Ci, Co, trans; };

__global__ __launch_bounds__(256)
void prepack_w(const float* __restrict__ w, unsigned* __restrict__ wp,
               PackP q, int total)
{
  const int id = (int)(blockIdx.x * blockDim.x + threadIdx.x);
  if (id >= total) return;
  const int i = id & 7, l = (id >> 3) & 15, g = (id >> 7) & 1;
  int rest = id >> 8;
  const int mt16 = rest % q.MT; rest /= q.MT;
  const int cc = rest % q.NC;   const int kr = rest / q.NC;
  const int co = mt16 * 16 + l;
  const int d0 = 2 * i;                              // A K-layout pair
  const int ci = cc * 32 + g * 16 + d0;
  float f0 = 0.f, f1 = 0.f;
  if (co < q.Co) {
    if (q.trans) {            // w[Ci,Co,kvol]
      f0 = w[((long)ci * q.Co + co) * q.kvol + kr];
      f1 = w[((long)(ci + 1) * q.Co + co) * q.kvol + kr];
    } else {                  // w[Co,Ci,kvol]
      f0 = w[((long)co * q.Ci + ci) * q.kvol + kr];
      f1 = w[((long)co * q.Ci + ci + 1) * q.kvol + kr];
    }
  }
  wp[id] = pk(f0, f1);
}

__global__ __launch_bounds__(256)
void f32_to_bf16(const float* __restrict__ in, ush* __restrict__ out, int n)
{
  const int i = (int)(blockIdx.x * blockDim.x + threadIdx.x);
  if (i < n) out[i] = (ush)((__float_as_uint(in[i]) + 0x8000u) >> 16);
}

// ---------------- host orchestration ----------------

extern "C" void kernel_launch(void* const* d_in, const int* in_sizes, int n_in,
                              void* d_out, int out_size, void* d_ws, size_t ws_size,
                              hipStream_t stream)
{
  const float* x = (const float*)d_in[0];
  auto W = [&](int i) { return (const float*)d_in[i]; };

  const long n0 = 20L * 24 * 24 * 24;   // 276480
  const long n1 = 10L * 12 * 12 * 12;   // 17280
  const long n2 = 5L  * 6 * 6 * 6;      // 1080
  const long n3 = 2L  * 3 * 3 * 3;      // 54

  // bf16 activation buffers, channel-minor [n][C]
  ush* A = (ush*)d_ws;
  ush* xb   = A;  A += n0;
  ush* t0   = A;  A += 32 * n0;        // scratch @ L0 (reused by up3a)
  ush* cat3 = A;  A += 64 * n0;        // [d1 | u3]
  ush* h3   = A;  A += 32 * n0;
  ush* d2   = A;  A += 32 * n1;
  ush* t1   = A;  A += 64 * n1;        // scratch @ L1 (reused by up2a)
  ush* cat2 = A;  A += 128 * n1;       // [d3 | u2]
  ush* h2   = A;  A += 64 * n1;
  ush* d4   = A;  A += 64 * n2;
  ush* t2   = A;  A += 128 * n2;       // scratch @ L2 (reused by up1a)
  ush* cat1 = A;  A += 256 * n2;       // [d5 | u1]
  ush* h1   = A;  A += 128 * n2;
  ush* d6   = A;  A += 128 * n3;
  ush* t3   = A;  A += 256 * n3;
  ush* d7   = A;  A += 256 * n3;

  unsigned* wpB = (unsigned*)(((size_t)A + 15) & ~(size_t)15);
  size_t wpOff = 0;

  auto conv = [&](const ush* xin, int ldIn, int cbIn,
                  const float* wraw, const float* b,
                  void* yout, int ldOut, int cbOut, int outF32,
                  int Ci, int Co, int Ti, int Di, int Hi, int Wi,
                  int kT, int kD, int kH, int kW, int s, int p,
                  int relu, int trans) {
    auto od = [&](int in, int k) { return trans ? (in - 1) * s - 2 * p + k
                                                : (in + 2 * p - k) / s + 1; };
    const int To = od(Ti, kT), Do = od(Di, kD), Ho = od(Hi, kH), Wo = od(Wi, kW);
    const int kvol = kT * kD * kH * kW;
    const int NC = Ci / 32;
    const int MT = 2 * ((Co + 31) / 32);
    // prepack weights into A-fragment order
    const int total = kvol * NC * MT * 256;
    unsigned* wp = wpB + wpOff; wpOff += (size_t)total;
    prepack_w<<<(total + 255) / 256, 256, 0, stream>>>(
        wraw, wp, PackP{kvol, NC, MT, Ci, Co, trans}, total);
    // conv
    ConvP P{Ci, Co, Ti, Di, Hi, Wi, To, Do, Ho, Wo, kT, kD, kH, kW, s, p, relu,
            ldIn, cbIn, ldOut, cbOut, NC, MT};
    const long Nout = (long)To * Do * Ho * Wo;
    const long waves = (long)((Co + 31) / 32) * ((Nout + 31) / 32);
    const int blocks = (int)((waves + 7) / 8);
    const size_t shmem = (size_t)kvol * sizeof(int);
    if (trans)       conv_bf16_wmma<true,  false><<<blocks, 256, shmem, stream>>>(xin, wp, b, yout, P);
    else if (outF32) conv_bf16_wmma<false, true ><<<blocks, 256, shmem, stream>>>(xin, wp, b, yout, P);
    else             conv_bf16_wmma<false, false><<<blocks, 256, shmem, stream>>>(xin, wp, b, yout, P);
  };

  // input f32 -> bf16
  f32_to_bf16<<<((int)n0 + 255) / 256, 256, 0, stream>>>(x, xb, (int)n0);

  // c1a (Ci=1 generic path): xb -> t0
  {
    ConvP P{1, 32, 20, 24, 24, 24, 20, 24, 24, 24, 3, 3, 3, 3, 1, 1, 1,
            1, 0, 32, 0, 0, 0};
    const long Nout = n0;
    const long waves = 2L * ((Nout + 15) / 16);
    const int blocks = (int)((waves + 7) / 8);
    conv_c1a<<<blocks, 256, 81 * sizeof(int), stream>>>(xb, W(1), W(2), t0, P);
  }

  // -------- encoder --------
  conv(t0,  32, 0, W(3),  W(4),  cat3, 64, 0, 0, 32, 32, 20,24,24,24, 3,3,3,3, 1,1, 1,0); // c1b -> d1 slice
  conv(cat3,64, 0, W(17), W(18), d2,   32, 0, 0, 32, 32, 20,24,24,24, 2,2,2,2, 2,0, 0,0); // ds1
  conv(d2,  32, 0, W(5),  W(6),  t1,   64, 0, 0, 32, 64, 10,12,12,12, 3,3,3,3, 1,1, 1,0); // c2a
  conv(t1,  64, 0, W(7),  W(8),  cat2,128, 0, 0, 64, 64, 10,12,12,12, 3,3,3,3, 1,1, 1,0); // c2b -> d3 slice
  conv(cat2,128,0, W(19), W(20), d4,   64, 0, 0, 64, 64, 10,12,12,12, 2,2,2,2, 2,0, 0,0); // ds2
  conv(d4,  64, 0, W(9),  W(10), t2,  128, 0, 0, 64,128, 5,6,6,6,     3,3,3,3, 1,1, 1,0); // c3a
  conv(t2, 128, 0, W(11), W(12), cat1,256, 0, 0, 128,128, 5,6,6,6,    3,3,3,3, 1,1, 1,0); // c3b -> d5 slice
  conv(cat1,256,0, W(21), W(22), d6,  128, 0, 0, 128,128, 5,6,6,6,    2,2,2,2, 2,0, 0,0); // ds3
  conv(d6, 128, 0, W(13), W(14), t3,  256, 0, 0, 128,256, 2,3,3,3,    3,3,3,3, 1,1, 1,0); // c4a
  conv(t3, 256, 0, W(15), W(16), d7,  256, 0, 0, 256,256, 2,3,3,3,    3,3,3,3, 1,1, 1,0); // c4b

  // -------- decoder (t-convs write directly into concat slices) --------
  conv(d7, 256, 0, W(37), W(38), cat1,256,128, 0, 256,128, 2,3,3,3,   5,4,4,4, 2,1, 0,1); // u1
  conv(cat1,256,0, W(23), W(24), t2,  128, 0, 0, 256,128, 5,6,6,6,    3,3,3,3, 1,1, 1,0); // up1a
  conv(t2, 128, 0, W(25), W(26), h1,  128, 0, 0, 128,128, 5,6,6,6,    3,3,3,3, 1,1, 1,0); // up1b
  conv(h1, 128, 0, W(39), W(40), cat2,128, 64, 0, 128, 64, 5,6,6,6,   4,4,4,4, 2,1, 0,1); // u2
  conv(cat2,128,0, W(27), W(28), t1,   64, 0, 0, 128, 64, 10,12,12,12,3,3,3,3, 1,1, 1,0); // up2a
  conv(t1,  64, 0, W(29), W(30), h2,   64, 0, 0, 64, 64, 10,12,12,12, 3,3,3,3, 1,1, 1,0); // up2b
  conv(h2,  64, 0, W(41), W(42), cat3, 64, 32, 0, 64, 32, 10,12,12,12,4,4,4,4, 2,1, 0,1); // u3
  conv(cat3,64, 0, W(31), W(32), t0,   32, 0, 0, 64, 32, 20,24,24,24, 3,3,3,3, 1,1, 1,0); // up3a
  conv(t0,  32, 0, W(33), W(34), h3,   32, 0, 0, 32, 32, 20,24,24,24, 3,3,3,3, 1,1, 1,0); // up3b

  // -------- 1x1 head: f32 output --------
  conv(h3,  32, 0, W(35), W(36), d_out, 0, 0, 1, 32, 1, 20,24,24,24,  1,1,1,1, 1,0, 0,0); // out
}
